// GlobalGraphPool_22359599743564
// MI455X (gfx1250) — matrix operand order repaired and checked by reading, not compiled, not run on hardware
//
#include <hip/hip_runtime.h>
#include <math.h>

// ---------------------------------------------------------------------------
// GlobalGraphPool for MI455X (gfx1250, wave32).
//   out[b] = concat(mean_pool, max_pool, attn_pool)   [B, 3*H], H=256
// Score GEMM (x @ W1) on v_wmma_f32_16x16x32_bf16 (f32 accumulate).
// x tiles staged global->LDS with double-buffered GLOBAL_LOAD_ASYNC_TO_LDS_B128
// (ASYNCcnt), overlapping HBM streaming with WMMA + pooling.
// One workgroup (256 thr = 8 waves) per graph segment; batch[] is sorted.
// ---------------------------------------------------------------------------

typedef __attribute__((ext_vector_type(16))) __bf16 v16bf;
typedef __attribute__((ext_vector_type(8)))  float  v8f;

constexpr int H = 256;

__device__ __forceinline__ void wait_asynccnt0() {
#if __has_builtin(__builtin_amdgcn_s_wait_asynccnt)
  __builtin_amdgcn_s_wait_asynccnt(0);
#else
  asm volatile("s_wait_asynccnt 0x0" ::: "memory");
#endif
}

// W1 (fp32, [k][n] row-major) -> W1T (bf16, [n][k]) so each lane's WMMA
// B-fragment (16 consecutive K values of one column n) is one contiguous
// 32-byte load. 128 KB total -> permanently L2 resident; compiler keeps all
// 16 fragments in VGPRs across the whole row loop.
__global__ __launch_bounds__(256) void ggp_prep_w1t(const float* __restrict__ W1,
                                                    __bf16* __restrict__ W1T) {
  int idx = blockIdx.x * blockDim.x + threadIdx.x;  // idx = n*H + k
  int n = idx >> 8;
  int k = idx & (H - 1);
  W1T[idx] = (__bf16)W1[k * H + n];
}

__global__ __launch_bounds__(256) void ggp_main(
    const float* __restrict__ x, const long long* __restrict__ batch,
    const float* __restrict__ b1, const float* __restrict__ W2,
    const float* __restrict__ b2, const __bf16* __restrict__ W1T,
    float* __restrict__ s_glob, float* __restrict__ out, int N) {
  const int b   = blockIdx.x;
  const int tid = threadIdx.x;

  __shared__ float xs[2][16][H + 4];  // double-buffered tile ring, padded rows
  __shared__ float s_tile[16];
  __shared__ float red[256];
  __shared__ float wbuf[256];
  __shared__ int   sh_start, sh_end;
  __shared__ float sh_max;

  // --- segment bounds via binary search on sorted int64 batch[] -------------
  if (tid == 0) {
    long long tgt = (long long)b;
    int lo = 0, hi = N;
    while (lo < hi) { int mid = (lo + hi) >> 1; if (batch[mid] < tgt) lo = mid + 1; else hi = mid; }
    sh_start = lo;
    hi = N; long long tgt1 = tgt + 1;
    while (lo < hi) { int mid = (lo + hi) >> 1; if (batch[mid] < tgt1) lo = mid + 1; else hi = mid; }
    sh_end = lo;
    sh_max = -INFINITY;
  }
  __syncthreads();

  const int start = sh_start, end = sh_end, cnt = end - start;
  float* outb = out + (size_t)b * (3 * H);

  if (cnt == 0) {                       // empty segment -> all zeros
    outb[tid] = 0.0f; outb[H + tid] = 0.0f; outb[2 * H + tid] = 0.0f;
    return;
  }

  const int wave  = tid >> 5;           // 0..7
  const int lane  = tid & 31;
  const int lhalf = lane >> 4;          // 0/1
  const int l16   = lane & 15;
  const int n0    = wave * 32;          // this wave owns columns [n0, n0+32)

  const uint64_t xbase = (uint64_t)(uintptr_t)x;

  // Issue one tile's async global->LDS copy: 4 x 16B per thread, coalesced.
  // Rows past the segment are CLAMPED to a valid row: the garbage they leave
  // in LDS only ever reaches pool loops bounded by `rows` and accumulator
  // rows whose s_tile entries are never read -- provably harmless.
  auto issue_tile = [&](int row0, int buf) {
    #pragma unroll
    for (int j = 0; j < 4; ++j) {
      int e = (tid << 2) + (j << 10);   // element in 16x256 tile
      int m = e >> 8, c = e & (H - 1);
      int row = row0 + m;
      if (row >= N) row = N - 1;        // keep address valid
      uint32_t voff = (uint32_t)row * (uint32_t)(H * 4) + (uint32_t)(c * 4);
      uint32_t lds  = (uint32_t)(uintptr_t)&xs[buf][m][c];
      asm volatile("global_load_async_to_lds_b128 %0, %1, %2"
                   :: "v"(lds), "v"(voff), "s"(xbase) : "memory");
    }
  };

  float col_sum = 0.0f;
  float col_max = -INFINITY;

  const int ntiles = (cnt + 15) >> 4;
  issue_tile(start, 0);                 // prefill pipeline

  // ===================== Pass A: scores + mean/max pools ====================
  for (int t = 0; t < ntiles; ++t) {
    const int row0 = start + t * 16;
    const int rows = min(16, end - row0);
    const float (*cur)[H + 4] = xs[t & 1];

    if (tid < 16) s_tile[tid] = 0.0f;   // prev tile readers done (end barrier)
    wait_asynccnt0();                   // tile t resident in LDS (this wave)
    __syncthreads();                    // ... and all other waves' portions
    if (t + 1 < ntiles) issue_tile(row0 + 16, (t + 1) & 1);  // overlap next

    // running mean/max pools: thread tid owns column tid
    for (int m = 0; m < rows; ++m) {
      float v = cur[m][tid];
      col_sum += v;
      col_max = fmaxf(col_max, v);
    }

    // --- WMMA: S[16 x 32cols] = Xtile[16x256] @ W1[:, n0:n0+32] -------------
    v8f acc0 = {}, acc1 = {};
    for (int kb = 0; kb < H; kb += 32) {
      // A fragment (16x32 bf16): lane holds row l16; K-halves split by lhalf
      const float* arow = &cur[l16][kb + 16 * lhalf];
      float4 c0 = *(const float4*)(arow);
      float4 c1 = *(const float4*)(arow + 4);
      float4 c2 = *(const float4*)(arow + 8);
      float4 c3 = *(const float4*)(arow + 12);
      v16bf a;
      a[0]=(__bf16)c0.x;  a[1]=(__bf16)c0.y;  a[2]=(__bf16)c0.z;  a[3]=(__bf16)c0.w;
      a[4]=(__bf16)c1.x;  a[5]=(__bf16)c1.y;  a[6]=(__bf16)c1.z;  a[7]=(__bf16)c1.w;
      a[8]=(__bf16)c2.x;  a[9]=(__bf16)c2.y;  a[10]=(__bf16)c2.z; a[11]=(__bf16)c2.w;
      a[12]=(__bf16)c3.x; a[13]=(__bf16)c3.y; a[14]=(__bf16)c3.z; a[15]=(__bf16)c3.w;

      // B fragments (32x16 bf16): lane = column, 16 contiguous K per lane
      const __bf16* bp0 = W1T + (size_t)(n0 + l16) * H + kb + 16 * lhalf;
      v16bf bf0 = *(const v16bf*)bp0;
      v16bf bf1 = *(const v16bf*)(bp0 + 16 * H);

      acc0 = __builtin_amdgcn_wmma_f32_16x16x32_bf16(false, a, false, bf0,
                                                     (short)0, acc0, false, false);
      acc1 = __builtin_amdgcn_wmma_f32_16x16x32_bf16(false, a, false, bf1,
                                                     (short)0, acc1, false, false);
    }

    // epilogue: s_row += sum_n tanh(acc + b1[n]) * W2[n]
    float part[8];
    #pragma unroll
    for (int r = 0; r < 8; ++r) part[r] = 0.0f;
    {
      int nA = n0 + l16;        float b1A = b1[nA], w2A = W2[nA];
      int nB = n0 + 16 + l16;   float b1B = b1[nB], w2B = W2[nB];
      #pragma unroll
      for (int r = 0; r < 8; ++r) {
        part[r] += tanhf(acc0[r] + b1A) * w2A;
        part[r] += tanhf(acc1[r] + b1B) * w2B;
      }
    }
    #pragma unroll
    for (int msk = 1; msk < 16; msk <<= 1) {
      #pragma unroll
      for (int r = 0; r < 8; ++r) part[r] += __shfl_xor(part[r], msk, 32);
    }
    if (l16 == 0) {
      #pragma unroll
      for (int r = 0; r < 8; ++r) atomicAdd(&s_tile[r + 8 * lhalf], part[r]);
    }
    __syncthreads();

    if (tid == 0) {             // finalize this tile's scores, track seg max
      float b2v = b2[0];
      float mx = sh_max;
      for (int m = 0; m < rows; ++m) {
        float sv = s_tile[m] + b2v;
        __hip_atomic_store(&s_glob[row0 + m], sv, __ATOMIC_RELAXED,
                           __HIP_MEMORY_SCOPE_AGENT);
        mx = fmaxf(mx, sv);
      }
      sh_max = mx;
    }
    __syncthreads();
  }

  // mean / max pool outputs (cnt > 0 here)
  outb[tid]     = col_sum / (float)cnt;
  outb[H + tid] = col_max;

  // ===================== softmax denom over the segment =====================
  const float mx = sh_max;
  float esum = 0.0f;
  for (int r = start + tid; r < end; r += 256) {
    float sv = __hip_atomic_load(&s_glob[r], __ATOMIC_RELAXED,
                                 __HIP_MEMORY_SCOPE_AGENT);
    esum += expf(sv - mx);
  }
  red[tid] = esum;
  __syncthreads();
  #pragma unroll
  for (int s = 128; s > 0; s >>= 1) {
    if (tid < s) red[tid] += red[tid + s];
    __syncthreads();
  }
  const float inv_denom = 1.0f / red[0];

  // ===================== Pass B: attention pool =============================
  float acc = 0.0f;
  for (int row0 = start; row0 < end; row0 += 256) {
    const int rows = min(256, end - row0);
    if (tid < rows) {
      float sv = __hip_atomic_load(&s_glob[row0 + tid], __ATOMIC_RELAXED,
                                   __HIP_MEMORY_SCOPE_AGENT);
      wbuf[tid] = expf(sv - mx) * inv_denom;
    }
    __syncthreads();
    for (int m = 0; m < rows; ++m)
      acc += wbuf[m] * x[(size_t)(row0 + m) * H + tid];   // coalesced
    __syncthreads();
  }
  outb[2 * H + tid] = acc;
}

// ---------------------------------------------------------------------------
extern "C" void kernel_launch(void* const* d_in, const int* in_sizes, int n_in,
                              void* d_out, int out_size, void* d_ws, size_t ws_size,
                              hipStream_t stream) {
  const float*     x     = (const float*)d_in[0];
  const long long* batch = (const long long*)d_in[1];   // int64 per reference
  const float*     W1    = (const float*)d_in[2];
  const float*     b1    = (const float*)d_in[3];
  const float*     W2    = (const float*)d_in[4];
  const float*     b2    = (const float*)d_in[5];

  const int N = in_sizes[0] / H;        // 500000
  const int B = out_size / (3 * H);     // 512

  // workspace: [ s_glob: N floats | W1T: H*H bf16 ]
  float* s_glob = (float*)d_ws;
  size_t off = ((size_t)N * sizeof(float) + 255) & ~(size_t)255;
  __bf16* W1T = (__bf16*)((char*)d_ws + off);

  ggp_prep_w1t<<<(H * H) / 256, 256, 0, stream>>>(W1, W1T);
  ggp_main<<<B, 256, 0, stream>>>(x, batch, b1, W2, b2, W1T, s_glob,
                                  (float*)d_out, N);
}
